// TransG_77017353552064
// MI455X (gfx1250) — matrix operand order repaired and checked by reading, not compiled, not run on hardware
//
#include <hip/hip_runtime.h>
#include <hip/hip_bf16.h>

typedef __attribute__((ext_vector_type(16))) _Float16 v16h;
typedef __attribute__((ext_vector_type(8)))  _Float16 h8;
typedef __attribute__((ext_vector_type(8)))  float    v8f;

union H16 { v16h v; h8 h[2]; };

__device__ __forceinline__ v8f vzero8f() {
    v8f z = {0.f,0.f,0.f,0.f,0.f,0.f,0.f,0.f};
    return z;
}

__device__ __forceinline__ float gelu_exact(float x) {
    return 0.5f * x * (1.0f + erff(x * 0.70710678118654752f));
}

// low 32 bits of a generic pointer to LDS == LDS byte offset (aperture in high bits)
__device__ __forceinline__ unsigned lds_lo32(const void* p) {
    return (unsigned)(size_t)p;
}

// async global->LDS copy of 16 bytes (wave-wide, tracked by ASYNCcnt)
__device__ __forceinline__ void async_b128(unsigned lds_off, const void* gptr) {
    asm volatile("global_load_async_to_lds_b128 %0, %1, off"
                 :: "v"(lds_off), "v"(gptr) : "memory");
}

// ---- problem constants ----
#define CB     16
#define CHW    4096          // 64*64
#define CC     512
#define CHID   2048
#define CHEADS 16
#define CHD    32
#define CN     64            // tokens per window
#define CROWS  65536         // B*H*W
#define CSCALE 0.17677669529663687f   // 32^-0.5

// =====================================================================
// Weight cast+transpose: f32 [K,N] row-major  ->  f16 [N,K] row-major
// =====================================================================
__global__ __launch_bounds__(256)
void wt_transpose(const float* __restrict__ in, _Float16* __restrict__ out,
                  int K, int N) {
    size_t i = (size_t)blockIdx.x * 256 + threadIdx.x;
    if (i >= (size_t)K * N) return;
    int k = (int)(i / N);
    int n = (int)(i % N);
    out[(size_t)n * K + k] = (_Float16)in[i];
}

// =====================================================================
// Relative-position bias table: tab[head][n][m] (f32, 16*64*64)
// =====================================================================
__global__ __launch_bounds__(256)
void build_bias(const float* __restrict__ rpb, float* __restrict__ tab) {
    int i = blockIdx.x * 256 + threadIdx.x;     // 0..65535
    int head = i >> 12;
    int n = (i >> 6) & 63;
    int m = i & 63;
    int rel = ((n >> 3) - (m >> 3) + 7) * 15 + ((n & 7) - (m & 7) + 7);
    tab[i] = rpb[rel * CHEADS + head];
}

// =====================================================================
// LayerNorm over C=512, optional roll(-4,-4)+window-partition remap.
// =====================================================================
template <bool WIN>
__global__ __launch_bounds__(256)
void ln_kernel(const float* __restrict__ x, const float* __restrict__ g,
               const float* __restrict__ b, _Float16* __restrict__ out) {
    const int r = blockIdx.x;            // 0..65535
    size_t src;
    if (WIN) {
        int win = r >> 6, tok = r & 63;
        int bb = win >> 6, wh = (win >> 3) & 7, ww = win & 7;
        int th = tok >> 3, tw = tok & 7;
        int h = ((wh << 3) + th + 4) & 63;     // roll(-4): out[h']=in[h'+4]
        int w = ((ww << 3) + tw + 4) & 63;
        src = ((size_t)bb * CHW + h * 64 + w) * CC;
    } else {
        src = (size_t)r * CC;
    }
    const int tid = threadIdx.x;
    float v0 = x[src + tid];
    float v1 = x[src + 256 + tid];
    __shared__ float s1[256], s2[256];
    s1[tid] = v0 + v1;
    s2[tid] = v0 * v0 + v1 * v1;
    __syncthreads();
    for (int off = 128; off > 0; off >>= 1) {
        if (tid < off) { s1[tid] += s1[tid + off]; s2[tid] += s2[tid + off]; }
        __syncthreads();
    }
    float mu  = s1[0] * (1.0f / 512.0f);
    float var = s2[0] * (1.0f / 512.0f) - mu * mu;
    float rs  = rsqrtf(var + 1e-5f);
    size_t dst = (size_t)r * CC;
    out[dst + tid]       = (_Float16)((v0 - mu) * rs * g[tid]       + b[tid]);
    out[dst + 256 + tid] = (_Float16)((v1 - mu) * rs * g[256 + tid] + b[256 + tid]);
}

// =====================================================================
// Epilogue functors
// =====================================================================
struct EpiQKV {
    const float* bias;
    _Float16 *q, *k, *v;
    __device__ __forceinline__ void operator()(int row, int col, float acc) const {
        float val = acc + bias[col];
        int which = col >> 9;
        int head  = (col >> 5) & 15;
        int d     = col & 31;
        int win   = row >> 6;
        int tok   = row & 63;
        size_t idx = (((size_t)win * CHEADS + head) * CN + tok) * CHD + d;
        if (which == 0)      q[idx] = (_Float16)(val * CSCALE);
        else if (which == 1) k[idx] = (_Float16)val;
        else                 v[idx] = (_Float16)val;
    }
};

struct EpiProj {   // + bias, window-reverse, roll(+4,+4), + shortcut -> xmid f32
    const float* bias;
    const float* shortcut;
    float* xmid;
    __device__ __forceinline__ void operator()(int row, int col, float acc) const {
        int win = row >> 6, tok = row & 63;
        int bb = win >> 6, wh = (win >> 3) & 7, ww = win & 7;
        int th = tok >> 3, tw = tok & 7;
        int h = ((wh << 3) + th + 4) & 63;
        int w = ((ww << 3) + tw + 4) & 63;
        size_t g = ((size_t)bb * CHW + h * 64 + w) * CC + col;
        xmid[g] = acc + bias[col] + shortcut[g];
    }
};

struct EpiGelu {   // + bias, gelu, store f16 (ld = 2048)
    const float* bias;
    _Float16* out;
    __device__ __forceinline__ void operator()(int row, int col, float acc) const {
        out[(size_t)row * CHID + col] = (_Float16)gelu_exact(acc + bias[col]);
    }
};

struct EpiFC2 {    // + bias + xmid -> final f32 output
    const float* bias;
    const float* xmid;
    float* out;
    __device__ __forceinline__ void operator()(int row, int col, float acc) const {
        size_t g = (size_t)row * CC + col;
        out[g] = acc + bias[col] + xmid[g];
    }
};

// =====================================================================
// WMMA GEMM:  C[M,N] = A[M,K] (f16, row-major) * Bt[N,K]^T (f16)
// block tile 128x128, 8 waves (4x2), wave tile 32x64, K-step 32.
// Global->LDS staging uses double-buffered GLOBAL_LOAD_ASYNC_TO_LDS_B128.
// =====================================================================
template <typename Epi>
__global__ __launch_bounds__(256)
void gemm128(const _Float16* __restrict__ A, const _Float16* __restrict__ Bt,
             int K, Epi epi) {
    __shared__ _Float16 As[2][128 * 40];
    __shared__ _Float16 Bs[2][128 * 40];

    const int tid  = threadIdx.x;
    const int bm   = blockIdx.y << 7;
    const int bn   = blockIdx.x << 7;
    const int wave = tid >> 5;
    const int lane = tid & 31;
    const int lr   = lane & 15;
    const int kh   = lane >> 4;
    const int wm   = (wave & 3) << 5;   // 0,32,64,96
    const int wn   = (wave >> 2) << 6;  // 0,64

    const int ldr = tid >> 1;           // staging row 0..127 (A and Bt)
    const int lds = (tid & 1) << 4;     // 0 or 16 halves

    v8f acc[2][4];
#pragma unroll
    for (int mi = 0; mi < 2; ++mi)
#pragma unroll
        for (int ni = 0; ni < 4; ++ni) acc[mi][ni] = vzero8f();

    const _Float16* Ag = A  + (size_t)(bm + ldr) * K + lds;
    const _Float16* Bg = Bt + (size_t)(bn + ldr) * K + lds;
    const unsigned aoff[2] = { lds_lo32(&As[0][ldr * 40 + lds]),
                               lds_lo32(&As[1][ldr * 40 + lds]) };
    const unsigned boff[2] = { lds_lo32(&Bs[0][ldr * 40 + lds]),
                               lds_lo32(&Bs[1][ldr * 40 + lds]) };

    auto issue = [&](int k0, int s) {
        async_b128(aoff[s],      Ag + k0);
        async_b128(aoff[s] + 16, Ag + k0 + 8);
        async_b128(boff[s],      Bg + k0);
        async_b128(boff[s] + 16, Bg + k0 + 8);
    };

    const int nk = K >> 5;
    issue(0, 0);

    for (int kt = 0; kt < nk; ++kt) {
        const int cur = kt & 1;
        if (kt + 1 < nk) {
            issue((kt + 1) << 5, cur ^ 1);
            // async loads complete in order: <=4 outstanding => stage `cur` landed
            asm volatile("s_wait_asynccnt 0x4" ::: "memory");
        } else {
            asm volatile("s_wait_asynccnt 0x0" ::: "memory");
        }
        __syncthreads();

        H16 af[2], bf[4];
#pragma unroll
        for (int mi = 0; mi < 2; ++mi) {
            const _Float16* p = &As[cur][(wm + mi * 16 + lr) * 40];
            af[mi].h[0] = *(const h8*)(p + kh * 8);         // K = kh*8 .. +7
            af[mi].h[1] = *(const h8*)(p + 16 + kh * 8);    // K = 16+kh*8 .. +7
        }
#pragma unroll
        for (int ni = 0; ni < 4; ++ni) {
            const _Float16* p = &Bs[cur][(wn + ni * 16 + lr) * 40];
            bf[ni].h[0] = *(const h8*)(p + kh * 8);         // K = kh*8 .. +7
            bf[ni].h[1] = *(const h8*)(p + 16 + kh * 8);    // K = 16+kh*8 .. +7
        }
#pragma unroll
        for (int mi = 0; mi < 2; ++mi)
#pragma unroll
            for (int ni = 0; ni < 4; ++ni)
                acc[mi][ni] = __builtin_amdgcn_wmma_f32_16x16x32_f16(
                    false, af[mi].v, false, bf[ni].v,
                    (short)0, acc[mi][ni], false, false);
        __syncthreads();   // all waves done reading before buffer is overwritten
    }

#pragma unroll
    for (int mi = 0; mi < 2; ++mi)
#pragma unroll
        for (int ni = 0; ni < 4; ++ni)
#pragma unroll
            for (int r = 0; r < 8; ++r) {
                int grow = bm + wm + mi * 16 + r + kh * 8;
                int gcol = bn + wn + ni * 16 + lr;
                epi(grow, gcol, acc[mi][ni][r]);
            }
}

// =====================================================================
// Attention: one wave per (window, head). 4 waves/block.
// S = Q*K^T (+bias+mask), softmax, O = P*V.  All via f16 WMMA.
// q/k/v layout: [win, head, tok(64), d(32)] f16.  out: [row, 512] f16.
// =====================================================================
__device__ __forceinline__ int regid(int v) { return v < 56 ? 0 : (v < 60 ? 1 : 2); }

__global__ __launch_bounds__(128)
void attn_kernel(const _Float16* __restrict__ qb, const _Float16* __restrict__ kb,
                 const _Float16* __restrict__ vb, const float* __restrict__ biasTab,
                 _Float16* __restrict__ outp) {
    __shared__ _Float16 Psh[4 * 64 * 72];
    __shared__ _Float16 Vsh[4 * 32 * 72];

    const int wave = threadIdx.x >> 5;
    const int lane = threadIdx.x & 31;
    const int lr   = lane & 15;
    const int kh   = lane >> 4;

    const int task = blockIdx.x * 4 + wave;      // 0..16383
    const int win  = task >> 4;
    const int head = task & 15;
    const int wh   = (win >> 3) & 7;
    const int ww   = win & 7;

    const size_t base = ((size_t)win * CHEADS + head) * CN * CHD;
    const _Float16* Q  = qb + base;
    const _Float16* Kp = kb + base;
    const _Float16* Vp = vb + base;
    const float*    bt = biasTab + ((size_t)head << 12);

    _Float16* Pw = Psh + wave * 64 * 72;
    _Float16* Vt = Vsh + wave * 32 * 72;

    // stage V transposed: Vt[d][tok]
#pragma unroll
    for (int t0 = 0; t0 < 2; ++t0) {
        int t = lane + t0 * 32;
        const _Float16* vr = Vp + (size_t)t * CHD;
#pragma unroll
        for (int d0 = 0; d0 < 32; d0 += 8) {
            h8 vv = *(const h8*)(vr + d0);
#pragma unroll
            for (int j = 0; j < 8; ++j) Vt[(d0 + j) * 72 + t] = vv[j];
        }
    }

    // B fragments of K^T + per-key mask region ids (loop invariant)
    H16 kf[4];
    int idm4[4];
#pragma unroll
    for (int ni = 0; ni < 4; ++ni) {
        int tm = ni * 16 + lr;
        idm4[ni] = regid(wh * 8 + (tm >> 3)) * 3 + regid(ww * 8 + (tm & 7));
        const _Float16* kr = Kp + (size_t)tm * CHD;
        kf[ni].h[0] = *(const h8*)(kr + kh * 8);
        kf[ni].h[1] = *(const h8*)(kr + 16 + kh * 8);
    }

    // ---- phase 1: S = Q K^T, bias+mask, softmax -> P in LDS ----
    for (int mi = 0; mi < 4; ++mi) {
        const _Float16* qr = Q + (size_t)(mi * 16 + lr) * CHD;
        H16 af;
        af.h[0] = *(const h8*)(qr + kh * 8);
        af.h[1] = *(const h8*)(qr + 16 + kh * 8);

        v8f s[4];
#pragma unroll
        for (int ni = 0; ni < 4; ++ni)
            s[ni] = __builtin_amdgcn_wmma_f32_16x16x32_f16(
                false, af.v, false, kf[ni].v, (short)0, vzero8f(), false, false);

#pragma unroll
        for (int r = 0; r < 8; ++r) {
            int tn = mi * 16 + r + kh * 8;          // query token
            int idn = regid(wh * 8 + (tn >> 3)) * 3 + regid(ww * 8 + (tn & 7));
            const float* brow = bt + tn * 64;
            float vals[4];
            float mx = -1e30f;
#pragma unroll
            for (int ni = 0; ni < 4; ++ni) {
                float bsv = brow[ni * 16 + lr];
                if (idn != idm4[ni]) bsv -= 100.0f;
                vals[ni] = s[ni][r] + bsv;
                mx = fmaxf(mx, vals[ni]);
            }
            for (int m = 1; m < 16; m <<= 1) mx = fmaxf(mx, __shfl_xor(mx, m, 32));
            float sum = 0.f;
#pragma unroll
            for (int ni = 0; ni < 4; ++ni) { vals[ni] = __expf(vals[ni] - mx); sum += vals[ni]; }
            for (int m = 1; m < 16; m <<= 1) sum += __shfl_xor(sum, m, 32);
            float rinv = 1.0f / sum;
#pragma unroll
            for (int ni = 0; ni < 4; ++ni)
                Pw[tn * 72 + ni * 16 + lr] = (_Float16)(vals[ni] * rinv);
        }
    }

    // ---- phase 2: O = P V ----
    H16 vf[2][2];
#pragma unroll
    for (int nd = 0; nd < 2; ++nd)
#pragma unroll
        for (int kt = 0; kt < 2; ++kt) {
            const _Float16* vr = Vt + (nd * 16 + lr) * 72 + kt * 32;
            vf[nd][kt].h[0] = *(const h8*)(vr + kh * 8);
            vf[nd][kt].h[1] = *(const h8*)(vr + 16 + kh * 8);
        }

    for (int mi = 0; mi < 4; ++mi) {
        v8f o[2] = {vzero8f(), vzero8f()};
#pragma unroll
        for (int kt = 0; kt < 2; ++kt) {
            const _Float16* pr = Pw + (mi * 16 + lr) * 72 + kt * 32;
            H16 af;
            af.h[0] = *(const h8*)(pr + kh * 8);
            af.h[1] = *(const h8*)(pr + 16 + kh * 8);
#pragma unroll
            for (int nd = 0; nd < 2; ++nd)
                o[nd] = __builtin_amdgcn_wmma_f32_16x16x32_f16(
                    false, af.v, false, vf[nd][kt].v, (short)0, o[nd], false, false);
        }
#pragma unroll
        for (int nd = 0; nd < 2; ++nd)
#pragma unroll
            for (int r = 0; r < 8; ++r) {
                int tok = mi * 16 + r + kh * 8;
                int d   = nd * 16 + lr;
                outp[((size_t)win * CN + tok) * CC + head * CHD + d] = (_Float16)o[nd][r];
            }
    }
}

// =====================================================================
// Depthwise 3x3 conv (SAME) + bias + GELU, NHWC, f16 in/out.
// =====================================================================
__global__ __launch_bounds__(256)
void dwconv_gelu(const _Float16* __restrict__ in, const float* __restrict__ w,
                 const float* __restrict__ bias, _Float16* __restrict__ out) {
    const int p  = blockIdx.x;          // b*4096 + h*64 + w
    const int bb = p >> 12;
    const int h  = (p >> 6) & 63;
    const int wd = p & 63;
    const int c0 = threadIdx.x * 8;

    float acc[8];
#pragma unroll
    for (int j = 0; j < 8; ++j) acc[j] = bias[c0 + j];

#pragma unroll
    for (int dh = 0; dh < 3; ++dh) {
        int hh = h + dh - 1;
        if (hh < 0 || hh > 63) continue;
#pragma unroll
        for (int dw = 0; dw < 3; ++dw) {
            int wn = wd + dw - 1;
            if (wn < 0 || wn > 63) continue;
            size_t pn = ((size_t)bb * CHW + hh * 64 + wn) * CHID + c0;
            h8 vv = *(const h8*)(in + pn);
            const float* wp = w + (dh * 3 + dw) * CHID + c0;
#pragma unroll
            for (int j = 0; j < 8; ++j) acc[j] += (float)vv[j] * wp[j];
        }
    }
    h8 r;
#pragma unroll
    for (int j = 0; j < 8; ++j) r[j] = (_Float16)gelu_exact(acc[j]);
    *(h8*)(out + (size_t)p * CHID + c0) = r;
}

// =====================================================================
// Launch
// =====================================================================
extern "C" void kernel_launch(void* const* d_in, const int* in_sizes, int n_in,
                              void* d_out, int out_size, void* d_ws, size_t ws_size,
                              hipStream_t stream) {
    const float* x      = (const float*)d_in[0];
    const float* n1g    = (const float*)d_in[1];
    const float* n1b    = (const float*)d_in[2];
    const float* qkv_w  = (const float*)d_in[3];
    const float* qkv_b  = (const float*)d_in[4];
    const float* rpb    = (const float*)d_in[5];
    const float* proj_w = (const float*)d_in[6];
    const float* proj_b = (const float*)d_in[7];
    const float* n2g    = (const float*)d_in[8];
    const float* n2b    = (const float*)d_in[9];
    const float* fc1_w  = (const float*)d_in[10];
    const float* fc1_b  = (const float*)d_in[11];
    const float* conv_w = (const float*)d_in[12];
    const float* conv_b = (const float*)d_in[13];
    const float* fc2_w  = (const float*)d_in[14];
    const float* fc2_b  = (const float*)d_in[15];
    float* out = (float*)d_out;

    char* ws = (char*)d_ws;
    const size_t S1 = (size_t)CROWS * CC * sizeof(_Float16);   // f16 [65536,512]
    _Float16* qB   = (_Float16*)(ws);
    _Float16* kB   = (_Float16*)(ws + S1);
    _Float16* vB   = (_Float16*)(ws + 2 * S1);
    _Float16* regX = (_Float16*)(ws + 3 * S1);   // xw -> attn_out -> ln2_out (reused)
    float*    xmid = (float*)   (ws + 4 * S1);
    char* p = ws + 4 * S1 + (size_t)CROWS * CC * sizeof(float);
    _Float16* h1 = (_Float16*)p; p += (size_t)CROWS * CHID * sizeof(_Float16);
    _Float16* h2 = (_Float16*)p; p += (size_t)CROWS * CHID * sizeof(_Float16);
    _Float16* wqkvT = (_Float16*)p; p += (size_t)CC * 3 * CC * sizeof(_Float16);
    _Float16* wprojT = (_Float16*)p; p += (size_t)CC * CC * sizeof(_Float16);
    _Float16* wfc1T = (_Float16*)p; p += (size_t)CC * CHID * sizeof(_Float16);
    _Float16* wfc2T = (_Float16*)p; p += (size_t)CHID * CC * sizeof(_Float16);
    float* biasTab = (float*)p;

    // weight cast + transpose to [N,K] f16; bias table
    wt_transpose<<<(512 * 1536 + 255) / 256, 256, 0, stream>>>(qkv_w, wqkvT, 512, 1536);
    wt_transpose<<<(512 * 512  + 255) / 256, 256, 0, stream>>>(proj_w, wprojT, 512, 512);
    wt_transpose<<<(512 * 2048 + 255) / 256, 256, 0, stream>>>(fc1_w, wfc1T, 512, 2048);
    wt_transpose<<<(2048 * 512 + 255) / 256, 256, 0, stream>>>(fc2_w, wfc2T, 2048, 512);
    build_bias<<<256, 256, 0, stream>>>(rpb, biasTab);

    // LN1 + roll + window-partition -> xw (regX)
    ln_kernel<true><<<CROWS, 256, 0, stream>>>(x, n1g, n1b, regX);

    // qkv GEMM -> q,k,v
    gemm128<EpiQKV><<<dim3(12, 512), 256, 0, stream>>>(regX, wqkvT, 512,
        EpiQKV{qkv_b, qB, kB, vB});

    // attention -> attn_out (regX, xw no longer needed)
    attn_kernel<<<4096, 128, 0, stream>>>(qB, kB, vB, biasTab, regX);

    // proj GEMM (+bias, window-reverse, roll, +shortcut) -> xmid
    gemm128<EpiProj><<<dim3(4, 512), 256, 0, stream>>>(regX, wprojT, 512,
        EpiProj{proj_b, x, xmid});

    // LN2 -> ln2_out (regX)
    ln_kernel<false><<<CROWS, 256, 0, stream>>>(xmid, n2g, n2b, regX);

    // fc1 GEMM + GELU -> h1
    gemm128<EpiGelu><<<dim3(16, 512), 256, 0, stream>>>(regX, wfc1T, 512,
        EpiGelu{fc1_b, h1});

    // depthwise conv + bias + GELU -> h2
    dwconv_gelu<<<CROWS, 256, 0, stream>>>(h1, conv_w, conv_b, h2);

    // fc2 GEMM + bias + residual -> out
    gemm128<EpiFC2><<<dim3(4, 512), 256, 0, stream>>>(h2, wfc2T, 2048,
        EpiFC2{fc2_b, xmid, out});
}
